// MoELayer_33973191311884
// MI455X (gfx1250) — matrix operand order repaired and checked by reading, not compiled, not run on hardware
//
#include <hip/hip_runtime.h>
#include <hip/hip_bf16.h>
#include <stdint.h>

// ---------------- problem constants (from reference) ----------------
#define T_TOK 4096          // B*S tokens
#define DDIM  1024          // model dim
#define FDIM  4096          // ffn dim
#define NEXP  8             // experts
// ---------------- tiling ----------------
#define BM    128           // slot-tile (M)
#define BN    64            // out-col tile (N)
#define BK    64            // K tile
#define AP    72            // LDS pitch (elems) for A tile  (mult of 8 -> 16B aligned rows)
#define BPI   72            // LDS pitch (elems) for transposed B tiles
#define SLOT_CAP 9216       // T*TOPK + E*(BM-1) rounded up

// ---------------- CDNA5 feature detection (device pass only) ----------------
#if defined(__AMDGCN__) && __has_builtin(__builtin_amdgcn_global_load_async_to_lds_b128) && \
    __has_builtin(__builtin_amdgcn_s_wait_asynccnt)
#define USE_ASYNC_COPY 1
#else
#define USE_ASYNC_COPY 0
#endif

#if defined(__AMDGCN__) && __has_builtin(__builtin_amdgcn_tensor_load_to_lds) && \
    __has_builtin(__builtin_amdgcn_s_wait_tensorcnt)
#define USE_TDM 1
#else
#define USE_TDM 0
#endif

typedef __attribute__((ext_vector_type(16))) __bf16 v16bf;
typedef __attribute__((ext_vector_type(8)))  float  v8f;
#if USE_TDM
typedef unsigned int u32x4 __attribute__((ext_vector_type(4)));
typedef int          i32x4 __attribute__((ext_vector_type(4)));
typedef int          i32x8 __attribute__((ext_vector_type(8)));
#endif
#if USE_ASYNC_COPY
// async-copy builtin wants: param0 = AS1 (global/"__device__") int4*, non-const;
// param1 = LDS side -> AS3 int4* (converts implicitly if the param is void*/generic)
typedef int i32x4v __attribute__((ext_vector_type(4)));
typedef __attribute__((address_space(1))) i32x4v* gas_ptr;
typedef __attribute__((address_space(3))) i32x4v* las_ptr;
#endif

__device__ __forceinline__ __bf16 f2bf(float f) {
  union { float f; uint32_t u; } a; a.f = f;
  uint32_t r = a.u + 0x7FFFu + ((a.u >> 16) & 1u);   // round-to-nearest-even
  union { uint16_t s; __bf16 b; } o; o.s = (uint16_t)(r >> 16);
  return o.b;
}

// silu(g)*v3 with fast reciprocal (bf16 output: v_rcp_f32 precision is plenty)
__device__ __forceinline__ float silu_mul(float g, float v3) {
  return g * __builtin_amdgcn_rcpf(1.0f + __expf(-g)) * v3;
}

// ---------------- phase 0: x fp32 -> bf16 ----------------
__global__ __launch_bounds__(256) void cvt_x_kernel(const float* __restrict__ x,
                                                    __bf16* __restrict__ xb) {
  size_t i = ((size_t)blockIdx.x * 256 + threadIdx.x) * 4;
  const float4 v = *(const float4*)(x + i);
  union { __bf16 b[4]; uint2 u; } o;
  o.b[0] = f2bf(v.x); o.b[1] = f2bf(v.y); o.b[2] = f2bf(v.z); o.b[3] = f2bf(v.w);
  *(uint2*)(xb + i) = o.u;
}

// ---------------- phase 1: router (one token per wave32) ----------------
__global__ __launch_bounds__(256) void router_kernel(const float* __restrict__ x,
                                                     const float* __restrict__ rw,
                                                     const float* __restrict__ rb,
                                                     int* __restrict__ topk_e,
                                                     float* __restrict__ topk_w,
                                                     int* __restrict__ counts) {
  const int t    = blockIdx.x * 8 + (threadIdx.x >> 5);
  const int lane = threadIdx.x & 31;
  const float* xr = x + (size_t)t * DDIM;

  float acc[NEXP];
#pragma unroll
  for (int e = 0; e < NEXP; ++e) acc[e] = 0.0f;

  for (int d = lane; d < DDIM; d += 32) {
    const float xv = xr[d];
#pragma unroll
    for (int e = 0; e < NEXP; ++e) acc[e] += xv * rw[e * DDIM + d];
  }
#pragma unroll
  for (int e = 0; e < NEXP; ++e)
    for (int off = 16; off > 0; off >>= 1) acc[e] += __shfl_down(acc[e], off, 32);

  if (lane == 0) {
    float lg[NEXP];
#pragma unroll
    for (int e = 0; e < NEXP; ++e) lg[e] = acc[e] + rb[e];
    int e0 = 0; float m0 = lg[0];
#pragma unroll
    for (int e = 1; e < NEXP; ++e) if (lg[e] > m0) { m0 = lg[e]; e0 = e; }
    int e1 = (e0 == 0) ? 1 : 0; float m1 = lg[e1];
#pragma unroll
    for (int e = 0; e < NEXP; ++e)
      if (e != e0 && lg[e] > m1) { m1 = lg[e]; e1 = e; }
    // renormalized top-2 softmax weights == softmax over the two logits
    const float w0 = 1.0f / (1.0f + __expf(m1 - m0));
    topk_e[2 * t + 0] = e0; topk_w[2 * t + 0] = w0;
    topk_e[2 * t + 1] = e1; topk_w[2 * t + 1] = 1.0f - w0;
    atomicAdd(&counts[e0], 1);
    atomicAdd(&counts[e1], 1);
  }
}

// ---------------- phase 2: tile-aligned exclusive prefix over counts ----------------
__global__ void prefix_kernel(const int* __restrict__ counts, int* __restrict__ offs) {
  if (threadIdx.x == 0 && blockIdx.x == 0) {
    int acc = 0;
    for (int e = 0; e < NEXP; ++e) {
      offs[e] = acc;
      acc += ((counts[e] + BM - 1) / BM) * BM;
    }
    offs[NEXP] = acc;
  }
}

// ---------------- phase 3: scatter token ids / routing scales to slots ----------------
__global__ __launch_bounds__(256) void scatter_kernel(const int* __restrict__ topk_e,
                                                      const float* __restrict__ topk_w,
                                                      const int* __restrict__ offs,
                                                      int* __restrict__ fill,
                                                      int* __restrict__ tok,
                                                      float* __restrict__ scl) {
  const int t = blockIdx.x * blockDim.x + threadIdx.x;
  if (t >= T_TOK) return;
#pragma unroll
  for (int k = 0; k < 2; ++k) {
    const int e    = topk_e[2 * t + k];
    const int pos  = atomicAdd(&fill[e], 1);
    const int slot = offs[e] + pos;
    tok[slot] = t;
    scl[slot] = topk_w[2 * t + k];
  }
}

// ---------------- phase 4: fused GEMM1/GEMM3 + SwiGLU -> h (bf16) ----------------
__global__ __launch_bounds__(256) void gemm13_kernel(const __bf16* __restrict__ xb,
                                                     const float* __restrict__ w1,
                                                     const float* __restrict__ w3,
                                                     const int* __restrict__ tok,
                                                     const int* __restrict__ offs,
                                                     __bf16* __restrict__ h) {
  const int e      = blockIdx.z;
  const int segEnd = offs[e + 1];
  const int sBase  = offs[e] + blockIdx.y * BM;
  if (sBase >= segEnd) return;                 // expert segment exhausted
  const int nBase = blockIdx.x * BN;

  __shared__ __align__(16) __bf16 sA[BM * AP];
  __shared__ __align__(16) __bf16 sB1[BN * BPI];
  __shared__ __align__(16) __bf16 sB3[BN * BPI];

  const int tid  = threadIdx.x;
  const int lane = tid & 31;
  const int wave = tid >> 5;
  const int wm   = (wave >> 1) * 32;   // wave M origin within tile (4 rows of waves)
  const int wn   = (wave & 1) * 32;    // wave N origin within tile (2 cols of waves)

  int tkn[4];
#pragma unroll
  for (int it = 0; it < 4; ++it) tkn[it] = tok[sBase + (tid >> 3) + it * 32];

  const size_t wOff = (size_t)e * DDIM * FDIM;
  const float* __restrict__ w1e = w1 + wOff;
  const float* __restrict__ w3e = w3 + wOff;

  v8f acc1[2][2] = {};
  v8f acc3[2][2] = {};

  for (int k0 = 0; k0 < DDIM; k0 += BK) {
    // A tile: 128x64 bf16, gathered token rows (xb already bf16 -> raw 16B copies)
#pragma unroll
    for (int it = 0; it < 4; ++it) {
      const int c   = tid + it * 256;
      const int row = c >> 3;
      const int kc  = (c & 7) * 8;
      const __bf16* src = xb + (size_t)tkn[it] * DDIM + k0 + kc;
#if USE_ASYNC_COPY
      // CDNA5 async copy: memory -> LDS without VGPR round-trip (ASYNCcnt)
      __builtin_amdgcn_global_load_async_to_lds_b128(
          (gas_ptr)(uintptr_t)src,
          (las_ptr)(uint32_t)(uintptr_t)&sA[row * AP + kc], 0, 0);
#else
      *(uint4*)(&sA[row * AP + kc]) = *(const uint4*)src;
#endif
    }
    // B tiles: 64(K)x64(N) fp32, convert to bf16 and store transposed [n][k]
#pragma unroll
    for (int it = 0; it < 4; ++it) {
      const int c  = tid + it * 256;
      const int kr = c >> 4;
      const int nc = (c & 15) * 4;
      const size_t gi = (size_t)(k0 + kr) * FDIM + nBase + nc;
      const float4 b1 = *(const float4*)(w1e + gi);
      const float4 b3 = *(const float4*)(w3e + gi);
      sB1[(nc + 0) * BPI + kr] = f2bf(b1.x);
      sB1[(nc + 1) * BPI + kr] = f2bf(b1.y);
      sB1[(nc + 2) * BPI + kr] = f2bf(b1.z);
      sB1[(nc + 3) * BPI + kr] = f2bf(b1.w);
      sB3[(nc + 0) * BPI + kr] = f2bf(b3.x);
      sB3[(nc + 1) * BPI + kr] = f2bf(b3.y);
      sB3[(nc + 2) * BPI + kr] = f2bf(b3.z);
      sB3[(nc + 3) * BPI + kr] = f2bf(b3.w);
    }
#if USE_ASYNC_COPY
    __builtin_amdgcn_s_wait_asynccnt(0);
#endif
    __syncthreads();

#pragma unroll
    for (int kk = 0; kk < BK; kk += 32) {
      const int ar = lane & 15;
      const int ak = kk + ((lane >> 4) << 3);   // A: K 0..7 / 8..15 per half-wave
      const int bk = kk + ((lane >> 4) << 3);   // B: K 0..7 / 8..15 per half-wave
      union { uint4 u[2]; v16bf v; } a[2], bf1[2], bf3[2];
#pragma unroll
      for (int i = 0; i < 2; ++i) {
        const int arow = (wm + i * 16 + ar) * AP;
        a[i].u[0] = *(const uint4*)(&sA[arow + ak]);
        a[i].u[1] = *(const uint4*)(&sA[arow + ak + 16]);
      }
#pragma unroll
      for (int j = 0; j < 2; ++j) {
        const int brow = (wn + j * 16 + ar) * BPI + bk;
        bf1[j].u[0] = *(const uint4*)(&sB1[brow]);
        bf1[j].u[1] = *(const uint4*)(&sB1[brow + 16]);
        bf3[j].u[0] = *(const uint4*)(&sB3[brow]);
        bf3[j].u[1] = *(const uint4*)(&sB3[brow + 16]);
      }
#pragma unroll
      for (int i = 0; i < 2; ++i)
#pragma unroll
        for (int j = 0; j < 2; ++j) {
          acc1[i][j] = __builtin_amdgcn_wmma_f32_16x16x32_bf16(
              false, a[i].v, false, bf1[j].v, (short)0, acc1[i][j], false, false);
          acc3[i][j] = __builtin_amdgcn_wmma_f32_16x16x32_bf16(
              false, a[i].v, false, bf3[j].v, (short)0, acc3[i][j], false, false);
        }
    }
    __syncthreads();
  }

  // epilogue: h = silu(x@w1) * (x@w3), stored bf16 slot-major
  const int col   = lane & 15;
  const int rbase = (lane >> 4) << 3;
#pragma unroll
  for (int i = 0; i < 2; ++i)
#pragma unroll
    for (int j = 0; j < 2; ++j)
#pragma unroll
      for (int r = 0; r < 8; ++r) {
        const float hv = silu_mul(acc1[i][j][r], acc3[i][j][r]);
        const int m = wm + i * 16 + rbase + r;
        const int f = nBase + wn + j * 16 + col;
        h[(size_t)(sBase + m) * FDIM + f] = f2bf(hv);
      }
}

// ---------------- phase 5: GEMM2 + routed scatter-add into out ----------------
__global__ __launch_bounds__(256) void gemm2_kernel(const __bf16* __restrict__ h,
                                                    const float* __restrict__ w2,
                                                    const int* __restrict__ tok,
                                                    const float* __restrict__ scl,
                                                    const int* __restrict__ offs,
                                                    float* __restrict__ out) {
  const int e      = blockIdx.z;
  const int segEnd = offs[e + 1];
  const int sBase  = offs[e] + blockIdx.y * BM;
  if (sBase >= segEnd) return;
  const int nBase = blockIdx.x * BN;           // D columns

  __shared__ __align__(16) __bf16 sA[BM * AP];
  __shared__ __align__(16) __bf16 sB[BN * BPI];

  const int tid  = threadIdx.x;
  const int lane = tid & 31;
  const int wave = tid >> 5;
  const int wm   = (wave >> 1) * 32;
  const int wn   = (wave & 1) * 32;

  const float* __restrict__ w2e = w2 + (size_t)e * FDIM * DDIM;

  v8f acc[2][2] = {};

#if USE_TDM
  // Tensor DMA descriptor (D#) for the A tile: uniform 2D tile of h,
  // tile = BK x BM elems (2B), row stride FDIM, LDS padded to AP pitch
  // (pad_interval code 4 = 128B rows, pad_amount code 3 = 16B -> 72-elem pitch).
  const uint32_t ldsA = (uint32_t)(uintptr_t)&sA[0];   // generic LDS ptr low 32 = LDS offset
  const uint64_t hBase = (uint64_t)(uintptr_t)(h + (size_t)sBase * FDIM);
  const i32x4 gz = {0, 0, 0, 0};
  i32x8 g1;
  g1[0] = (1 << 16) | (1 << 20) | (4 << 22) | (3 << 25);  // data_size=2B|pad_en|int128B|pad16B
  g1[1] = (int)((unsigned)BK << 16);                      // tensor_dim0 lo16 at [63:48]
  g1[2] = (int)((unsigned)BM << 16);                      // dim0 hi | tensor_dim1 lo16
  g1[3] = (int)((unsigned)BK << 16);                      // dim1 hi | tile_dim0
  g1[4] = BM;                                             // tile_dim1 | tile_dim2=0
  g1[5] = FDIM;                                           // tensor_dim0_stride lo32
  g1[6] = 0;
  g1[7] = 0;
#endif

  for (int k0 = 0; k0 < FDIM; k0 += BK) {
#if USE_TDM
    if (wave == 0) {
      const uint64_t ga = hBase + (size_t)k0 * 2;
      const u32x4 g0 = {1u,                                // count=1 (valid user D#)
                        ldsA,                              // lds_addr
                        (uint32_t)ga,                      // global_addr lo
                        (uint32_t)(ga >> 32) | (2u << 30)};// global_addr hi | type=2
#if __clang_major__ >= 23
      const i32x8 gz8 = {0, 0, 0, 0, 0, 0, 0, 0};
      __builtin_amdgcn_tensor_load_to_lds(g0, g1, gz, gz, gz8, 0);
#else
      __builtin_amdgcn_tensor_load_to_lds(g0, g1, gz, gz, 0);
#endif
    }
#else
    // A tile: h rows are contiguous slots (no gather needed)
#pragma unroll
    for (int it = 0; it < 4; ++it) {
      const int c   = tid + it * 256;
      const int row = c >> 3;
      const int kc  = (c & 7) * 8;
      const uint4 v = *(const uint4*)(h + (size_t)(sBase + row) * FDIM + k0 + kc);
      *(uint4*)(&sA[row * AP + kc]) = v;
    }
#endif
    // B tile: w2[e] rows k (F), cols d (D) -> bf16 transposed [d][k]
#pragma unroll
    for (int it = 0; it < 4; ++it) {
      const int c  = tid + it * 256;
      const int kr = c >> 4;
      const int nc = (c & 15) * 4;
      const float4 b = *(const float4*)(w2e + (size_t)(k0 + kr) * DDIM + nBase + nc);
      sB[(nc + 0) * BPI + kr] = f2bf(b.x);
      sB[(nc + 1) * BPI + kr] = f2bf(b.y);
      sB[(nc + 2) * BPI + kr] = f2bf(b.z);
      sB[(nc + 3) * BPI + kr] = f2bf(b.w);
    }
#if USE_TDM
    if (wave == 0) __builtin_amdgcn_s_wait_tensorcnt(0);
#endif
    __syncthreads();

#pragma unroll
    for (int kk = 0; kk < BK; kk += 32) {
      const int ar = lane & 15;
      const int ak = kk + ((lane >> 4) << 3);
      const int bk = kk + ((lane >> 4) << 3);
      union { uint4 u[2]; v16bf v; } a[2], bf[2];
#pragma unroll
      for (int i = 0; i < 2; ++i) {
        const int arow = (wm + i * 16 + ar) * AP;
        a[i].u[0] = *(const uint4*)(&sA[arow + ak]);
        a[i].u[1] = *(const uint4*)(&sA[arow + ak + 16]);
      }
#pragma unroll
      for (int j = 0; j < 2; ++j) {
        const int brow = (wn + j * 16 + ar) * BPI + bk;
        bf[j].u[0] = *(const uint4*)(&sB[brow]);
        bf[j].u[1] = *(const uint4*)(&sB[brow + 16]);
      }
#pragma unroll
      for (int i = 0; i < 2; ++i)
#pragma unroll
        for (int j = 0; j < 2; ++j)
          acc[i][j] = __builtin_amdgcn_wmma_f32_16x16x32_bf16(
              false, a[i].v, false, bf[j].v, (short)0, acc[i][j], false, false);
    }
    __syncthreads();
  }

  // epilogue: out[token, d] += scale * acc  (padding slots have scale==0, token==0)
  const int col   = lane & 15;
  const int rbase = (lane >> 4) << 3;
#pragma unroll
  for (int i = 0; i < 2; ++i)
#pragma unroll
    for (int j = 0; j < 2; ++j)
#pragma unroll
      for (int r = 0; r < 8; ++r) {
        const int m    = wm + i * 16 + rbase + r;
        const int slot = sBase + m;
        const int d    = nBase + wn + j * 16 + col;
        const float v  = acc[i][j][r] * scl[slot];
        atomicAdd(out + (size_t)tok[slot] * DDIM + d, v);
      }
}

// ---------------- host launcher ----------------
extern "C" void kernel_launch(void* const* d_in, const int* in_sizes, int n_in,
                              void* d_out, int out_size, void* d_ws, size_t ws_size,
                              hipStream_t stream) {
  (void)in_sizes; (void)n_in; (void)ws_size;
  const float* x  = (const float*)d_in[0];
  const float* rw = (const float*)d_in[1];
  const float* rb = (const float*)d_in[2];
  const float* w1 = (const float*)d_in[3];
  const float* w2 = (const float*)d_in[4];
  const float* w3 = (const float*)d_in[5];
  float* out = (float*)d_out;

  // workspace bump allocator (256B aligned)
  char* p = (char*)d_ws;
  auto bump = [&p](size_t bytes) -> void* {
    void* r = (void*)p;
    p += (bytes + 255) & ~(size_t)255;
    return r;
  };
  __bf16* xb   = (__bf16*)bump((size_t)T_TOK * DDIM * 2);            // 8 MB
  __bf16* hbuf = (__bf16*)bump((size_t)SLOT_CAP * FDIM * 2);         // 72 MB
  int*    tok  = (int*)   bump((size_t)SLOT_CAP * 4);
  float*  scl  = (float*) bump((size_t)SLOT_CAP * 4);
  int*    tke  = (int*)   bump((size_t)T_TOK * 2 * 4);
  float*  tkw  = (float*) bump((size_t)T_TOK * 2 * 4);
  int*    meta = (int*)   bump((size_t)(3 * NEXP + 1) * 4);
  int* counts = meta;
  int* fill   = meta + NEXP;
  int* offs   = meta + 2 * NEXP;

  (void)hipMemsetAsync(out, 0, (size_t)out_size * sizeof(float), stream);
  (void)hipMemsetAsync(meta, 0, (size_t)(3 * NEXP + 1) * 4, stream);
  (void)hipMemsetAsync(tok, 0, (size_t)SLOT_CAP * 4, stream);   // padding slots -> token 0
  (void)hipMemsetAsync(scl, 0, (size_t)SLOT_CAP * 4, stream);   // padding slots -> scale 0

  cvt_x_kernel<<<(T_TOK * DDIM) / (256 * 4), 256, 0, stream>>>(x, xb);
  router_kernel<<<T_TOK / 8, 256, 0, stream>>>(x, rw, rb, tke, tkw, counts);
  prefix_kernel<<<1, 32, 0, stream>>>(counts, offs);
  scatter_kernel<<<T_TOK / 256, 256, 0, stream>>>(tke, tkw, offs, fill, tok, scl);

  gemm13_kernel<<<dim3(FDIM / BN, T_TOK / BM, NEXP), 256, 0, stream>>>(
      xb, w1, w3, tok, offs, hbuf);
  gemm2_kernel<<<dim3(DDIM / BN, T_TOK / BM, NEXP), 256, 0, stream>>>(
      hbuf, w2, tok, scl, offs, out);
}
